// TransformerGNN_41979010351139
// MI455X (gfx1250) — matrix operand order repaired and checked by reading, not compiled, not run on hardware
//
#include <hip/hip_runtime.h>
#include <hip/hip_bf16.h>

#define NNODES 50000
#define NEDGES 800000

typedef __attribute__((ext_vector_type(16))) __bf16 v16bf;
typedef __attribute__((ext_vector_type(8)))  float  v8f;

// ---------------- helpers ----------------

__device__ __forceinline__ unsigned short f32_to_bf16_rne(float f) {
  unsigned int u = __float_as_uint(f);
  u += 0x7FFFu + ((u >> 16) & 1u);   // round to nearest even
  return (unsigned short)(u >> 16);
}

// order-preserving encode of float into uint for atomicMax
__device__ __forceinline__ unsigned int enc_ord(float f) {
  unsigned int u = __float_as_uint(f);
  return (u >> 31) ? ~u : (u | 0x80000000u);
}
__device__ __forceinline__ float dec_ord(unsigned int u) {
  u = (u >> 31) ? (u ^ 0x80000000u) : ~u;
  return __uint_as_float(u);
}

// ---------------- prep kernels ----------------

__global__ void convert_f32_to_bf16(const float* __restrict__ in,
                                    unsigned short* __restrict__ out, int n) {
  int i = blockIdx.x * blockDim.x + threadIdx.x;
  if (i < n) out[i] = f32_to_bf16_rne(in[i]);
}

// Pack 4 [fin, fout_each] fp32 weight matrices into one transposed bf16
// matrix Wt[4*fout_each][fin], and concat biases.
__global__ void pack_weights(const float* __restrict__ Wq, const float* __restrict__ Wk,
                             const float* __restrict__ Wv, const float* __restrict__ Ws,
                             const float* __restrict__ bq, const float* __restrict__ bk,
                             const float* __restrict__ bv, const float* __restrict__ bs,
                             unsigned short* __restrict__ Wt, float* __restrict__ bcat,
                             int fin, int fout_each) {
  int total = fin * fout_each * 4;
  int i = blockIdx.x * blockDim.x + threadIdx.x;
  if (i >= total) return;
  int nout = i / fin;
  int k    = i % fin;
  int seg  = nout / fout_each;
  int n    = nout % fout_each;
  const float* W = (seg == 0) ? Wq : (seg == 1) ? Wk : (seg == 2) ? Wv : Ws;
  Wt[(size_t)nout * fin + k] = f32_to_bf16_rne(W[(size_t)k * fout_each + n]);
  if (k == 0) {
    const float* B = (seg == 0) ? bq : (seg == 1) ? bk : (seg == 2) ? bv : bs;
    bcat[nout] = B[n];
  }
}

// ---------------- WMMA GEMM: Out[M,NOUT] = Xb[M,128] * Wt^T + bias ----------------
// Xb: bf16 row-major [M,128]; Wt: bf16 [NOUT,128] (i.e. B stored column-major).
// One wave computes one 16x16 tile; 4 waves/block; K=128 unrolled (4 WMMAs).
// NOUT is a template arg so all output addressing folds into immediate offsets.

template <int NOUT>
__global__ void __launch_bounds__(128)
gemm_bf16_wmma(const unsigned short* __restrict__ Xb,
               const unsigned short* __restrict__ Wt,
               const float* __restrict__ bias,
               float* __restrict__ Out) {
  const int lane  = threadIdx.x & 31;
  const int wave  = threadIdx.x >> 5;
  const int half  = lane >> 4;   // 0: lanes 0-15, 1: lanes 16-31
  const int lm    = lane & 15;
  const int mtile = blockIdx.x;
  const int ntile = blockIdx.y * 4 + wave;

  const unsigned short* arow = Xb + (size_t)(mtile * 16 + lm) * 128;  // A row (M = lm)
  const unsigned short* brow = Wt + (size_t)(ntile * 16 + lm) * 128;  // B col (N = lm)

  v8f c;
  const float bv = bias[ntile * 16 + lm];   // C/D: N = lane%16
#pragma unroll
  for (int r = 0; r < 8; ++r) c[r] = bv;

#pragma unroll
  for (int kb = 0; kb < 128; kb += 32) {
    union { v16bf v; uint4 q[2]; } A, B;
    // 16-bit A 16x32 layout: half0 elems = K{0..7,16..23}, half1 = K{8..15,24..31}
    A.q[0] = *(const uint4*)(arow + kb + half * 8);
    A.q[1] = *(const uint4*)(arow + kb + 16 + half * 8);
    // 16-bit B 32x16 layout: half0 elems = K{0..7,16..23}, half1 = K{8..15,24..31}
    B.q[0] = *(const uint4*)(brow + kb + half * 8);
    B.q[1] = *(const uint4*)(brow + kb + 16 + half * 8);
    c = __builtin_amdgcn_wmma_f32_16x16x32_bf16(false, A.v, false, B.v,
                                                (short)0, c, false, false);
  }

  // D layout: VGPR r, lane -> M = r + half*8, N = lane%16
  float* orow = Out + (size_t)(mtile * 16 + half * 8) * NOUT + ntile * 16 + lm;
#pragma unroll
  for (int r = 0; r < 8; ++r) orow[r * NOUT] = c[r];
}

// ---------------- edge kernels (one wave32 per edge) ----------------
// CT = heads*C channels total; P = CT/32 floats per lane; G = 32/H lanes per head.

template <int H, int CT>
__global__ void __launch_bounds__(256)
edge_logits(const int* __restrict__ srcs, const int* __restrict__ dsts,
            const float* __restrict__ qkvs, int rowstride, int qoff, int koff,
            float* __restrict__ logits, unsigned int* __restrict__ menc, int ne) {
  constexpr int P = CT / 32;
  constexpr int G = 32 / H;
  constexpr int C = CT / H;
  const int lane = threadIdx.x & 31;
  const int e = blockIdx.x * (blockDim.x >> 5) + (threadIdx.x >> 5);
  if (e >= ne) return;
  const int s = srcs[e], d = dsts[e];
  const float* qp = qkvs + (size_t)d * rowstride + qoff + lane * P;
  const float* kp = qkvs + (size_t)s * rowstride + koff + lane * P;
  float dot;
  if constexpr (P == 4) {
    float4 q = *(const float4*)qp, k = *(const float4*)kp;
    dot = q.x * k.x + q.y * k.y + q.z * k.z + q.w * k.w;
  } else {
    float2 q = *(const float2*)qp, k = *(const float2*)kp;
    dot = q.x * k.x + q.y * k.y;
  }
#pragma unroll
  for (int off = G >> 1; off > 0; off >>= 1) dot += __shfl_xor(dot, off, 32);
  if ((lane & (G - 1)) == 0) {
    const int h = lane / G;
    const float lg = dot * rsqrtf((float)C);
    logits[(size_t)e * H + h] = lg;
    atomicMax(&menc[(size_t)d * H + h], enc_ord(lg));
  }
}

template <int H, int CT>
__global__ void __launch_bounds__(256)
edge_accum(const int* __restrict__ srcs, const int* __restrict__ dsts,
           const float* __restrict__ qkvs, int rowstride, int voff,
           const float* __restrict__ logits, const unsigned int* __restrict__ menc,
           float* __restrict__ denom, float* __restrict__ msg, int ne) {
  constexpr int P = CT / 32;
  constexpr int G = 32 / H;
  const int lane = threadIdx.x & 31;
  const int e = blockIdx.x * (blockDim.x >> 5) + (threadIdx.x >> 5);
  if (e >= ne) return;
  const int s = srcs[e], d = dsts[e];
  const int h = lane / G;
  const float lg = logits[(size_t)e * H + h];
  const float m  = dec_ord(menc[(size_t)d * H + h]);
  const float a  = __expf(lg - m);
  const float* vp = qkvs + (size_t)s * rowstride + voff + lane * P;
  float* mp = msg + (size_t)d * CT + lane * P;
#pragma unroll
  for (int j = 0; j < P; ++j) atomicAdd(mp + j, a * vp[j]);
  if ((lane & (G - 1)) == 0) atomicAdd(&denom[(size_t)d * H + h], a);
}

// ---------------- finalize kernels ----------------

__global__ void finalize1(const float* __restrict__ msg, const float* __restrict__ denom,
                          const float* __restrict__ qkvs1,
                          unsigned short* __restrict__ h_bf, int total) {
  int i = blockIdx.x * blockDim.x + threadIdx.x;
  if (i >= total) return;
  const int n = i >> 7, c = i & 127, h = c >> 5;
  const float dn = denom[n * 4 + h];
  const float mv = (dn > 0.f) ? msg[i] / dn : 0.f;
  float v = mv + qkvs1[(size_t)n * 512 + 384 + c];   // + skip (x @ Ws + bs)
  v = v > 0.f ? v : 0.f;                             // ReLU
  h_bf[i] = f32_to_bf16_rne(v);
}

__global__ void finalize2(const float* __restrict__ msg, const float* __restrict__ denom,
                          const float* __restrict__ qkvs2,
                          float* __restrict__ out, int total) {
  int i = blockIdx.x * blockDim.x + threadIdx.x;
  if (i >= total) return;
  const int n = i >> 6, c = i & 63;
  const float dn = denom[n];
  const float mv = (dn > 0.f) ? msg[i] / dn : 0.f;
  out[i] = mv + qkvs2[(size_t)n * 256 + 192 + c];    // heads=1, mean == identity
}

// ---------------- host launcher ----------------

extern "C" void kernel_launch(void* const* d_in, const int* in_sizes, int n_in,
                              void* d_out, int out_size, void* d_ws, size_t ws_size,
                              hipStream_t stream) {
  const float* x   = (const float*)d_in[0];
  const int*   ei  = (const int*)d_in[1];
  const float* W1q = (const float*)d_in[2];  const float* b1q = (const float*)d_in[3];
  const float* W1k = (const float*)d_in[4];  const float* b1k = (const float*)d_in[5];
  const float* W1v = (const float*)d_in[6];  const float* b1v = (const float*)d_in[7];
  const float* W1s = (const float*)d_in[8];  const float* b1s = (const float*)d_in[9];
  const float* W2q = (const float*)d_in[10]; const float* b2q = (const float*)d_in[11];
  const float* W2k = (const float*)d_in[12]; const float* b2k = (const float*)d_in[13];
  const float* W2v = (const float*)d_in[14]; const float* b2v = (const float*)d_in[15];
  const float* W2s = (const float*)d_in[16]; const float* b2s = (const float*)d_in[17];
  const int* srcs = ei;
  const int* dsts = ei + NEDGES;

  char* ws = (char*)d_ws;
  size_t off = 0;
  auto carve = [&](size_t bytes) { size_t o = off; off += (bytes + 255) & ~(size_t)255; return o; };
  const size_t off_xbf   = carve((size_t)NNODES * 128 * 2);   // x as bf16
  const size_t off_wt1   = carve((size_t)512 * 128 * 2);      // packed W1 (bf16, transposed)
  const size_t off_b1    = carve((size_t)512 * 4);
  const size_t off_wt2   = carve((size_t)256 * 128 * 2);
  const size_t off_b2    = carve((size_t)256 * 4);
  const size_t off_qkvs  = carve((size_t)NNODES * 512 * 4);   // layer1 q|k|v|s; layer2 aliases
  const size_t off_log   = carve((size_t)NEDGES * 4 * 4);     // logits (layer2 aliases)
  const size_t off_m     = carve((size_t)NNODES * 4 * 4);     // seg-max (ordered uint)
  const size_t off_den   = carve((size_t)NNODES * 4 * 4);     // denom
  const size_t off_msg   = carve((size_t)NNODES * 128 * 4);   // numerator (layer2 aliases)
  const size_t off_hbf   = carve((size_t)NNODES * 128 * 2);   // layer1 output as bf16
  (void)ws_size;

  unsigned short* x_bf  = (unsigned short*)(ws + off_xbf);
  unsigned short* Wt1   = (unsigned short*)(ws + off_wt1);
  float*          bc1   = (float*)(ws + off_b1);
  unsigned short* Wt2   = (unsigned short*)(ws + off_wt2);
  float*          bc2   = (float*)(ws + off_b2);
  float*          qkvs  = (float*)(ws + off_qkvs);
  float*          logit = (float*)(ws + off_log);
  unsigned int*   menc  = (unsigned int*)(ws + off_m);
  float*          den   = (float*)(ws + off_den);
  float*          msg   = (float*)(ws + off_msg);
  unsigned short* h_bf  = (unsigned short*)(ws + off_hbf);

  // ---- prep: convert x, pack weights ----
  convert_f32_to_bf16<<<(NNODES * 128 + 255) / 256, 256, 0, stream>>>(x, x_bf, NNODES * 128);
  pack_weights<<<(512 * 128 + 255) / 256, 256, 0, stream>>>(
      W1q, W1k, W1v, W1s, b1q, b1k, b1v, b1s, Wt1, bc1, 128, 128);
  pack_weights<<<(256 * 128 + 255) / 256, 256, 0, stream>>>(
      W2q, W2k, W2v, W2s, b2q, b2k, b2v, b2s, Wt2, bc2, 128, 64);

  // ---- layer 1: fused q|k|v|s projection via WMMA (N=512) ----
  gemm_bf16_wmma<512><<<dim3(NNODES / 16, 512 / 16 / 4), 128, 0, stream>>>(x_bf, Wt1, bc1, qkvs);

  hipMemsetAsync(menc, 0, (size_t)NNODES * 4 * 4, stream);   // encoded floor (< any finite)
  hipMemsetAsync(den,  0, (size_t)NNODES * 4 * 4, stream);
  hipMemsetAsync(msg,  0, (size_t)NNODES * 128 * 4, stream);

  const int eblocks = (NEDGES + 7) / 8;                      // 8 waves (edges) per 256-thr block
  edge_logits<4, 128><<<eblocks, 256, 0, stream>>>(srcs, dsts, qkvs, 512, 0, 128,
                                                   logit, menc, NEDGES);
  edge_accum<4, 128><<<eblocks, 256, 0, stream>>>(srcs, dsts, qkvs, 512, 256,
                                                  logit, menc, den, msg, NEDGES);
  finalize1<<<(NNODES * 128 + 255) / 256, 256, 0, stream>>>(msg, den, qkvs, h_bf, NNODES * 128);

  // ---- layer 2: fused projection (N=256), heads=1, C=64 ----
  gemm_bf16_wmma<256><<<dim3(NNODES / 16, 256 / 16 / 4), 128, 0, stream>>>(h_bf, Wt2, bc2, qkvs);

  hipMemsetAsync(menc, 0, (size_t)NNODES * 4, stream);
  hipMemsetAsync(den,  0, (size_t)NNODES * 4, stream);
  hipMemsetAsync(msg,  0, (size_t)NNODES * 64 * 4, stream);

  edge_logits<1, 64><<<eblocks, 256, 0, stream>>>(srcs, dsts, qkvs, 256, 0, 64,
                                                  logit, menc, NEDGES);
  edge_accum<1, 64><<<eblocks, 256, 0, stream>>>(srcs, dsts, qkvs, 256, 128,
                                                 logit, menc, den, msg, NEDGES);
  finalize2<<<(NNODES * 64 + 255) / 256, 256, 0, stream>>>(msg, den, qkvs,
                                                           (float*)d_out, NNODES * 64);
}
